// LogitLinear_19816979104468
// MI455X (gfx1250) — matrix operand order repaired and checked by reading, not compiled, not run on hardware
//
#include <hip/hip_runtime.h>
#include <hip/hip_bf16.h>
#include <math.h>

typedef __attribute__((ext_vector_type(16))) _Float16 v16h;
typedef __attribute__((ext_vector_type(8)))  _Float16 v8h;
typedef __attribute__((ext_vector_type(4)))  _Float16 v4h;
typedef __attribute__((ext_vector_type(8)))  float    v8f;
typedef __attribute__((ext_vector_type(4)))  float    v4f;

static constexpr int IN  = 4096;   // in_features (K)
static constexpr int OUT = 4096;   // out_features (N)

// ---------------------------------------------------------------------------
// Kernel 1: convert x (f32) -> f16, 8 elements/thread
// ---------------------------------------------------------------------------
__global__ __launch_bounds__(256) void k_cvt_x(const float* __restrict__ x,
                                               _Float16* __restrict__ xh,
                                               int total) {
  int idx = (blockIdx.x * blockDim.x + threadIdx.x) * 8;
  if (idx >= total) return;
  v4f a = *(const v4f*)(x + idx);
  v4f b = *(const v4f*)(x + idx + 4);
  v8h o;
  o[0] = (_Float16)a[0]; o[1] = (_Float16)a[1];
  o[2] = (_Float16)a[2]; o[3] = (_Float16)a[3];
  o[4] = (_Float16)b[0]; o[5] = (_Float16)b[1];
  o[6] = (_Float16)b[2]; o[7] = (_Float16)b[3];
  *(v8h*)(xh + idx) = o;
}

// ---------------------------------------------------------------------------
// Kernel 2: W_logits (3, OUT, IN) -> w_mean, w_var (f16), 4 elements/thread.
// vals = [-1,0,1] => mean = p2-p0, E[w^2] = p2+p0, var = E[w^2]-mean^2
// ---------------------------------------------------------------------------
__global__ __launch_bounds__(256) void k_prep_w(const float* __restrict__ wl,
                                                _Float16* __restrict__ wm,
                                                _Float16* __restrict__ wv,
                                                int total) {
  int idx = (blockIdx.x * blockDim.x + threadIdx.x) * 4;
  if (idx >= total) return;
  v4f l0 = *(const v4f*)(wl + idx);                     // logits for value -1
  v4f l1 = *(const v4f*)(wl + (size_t)total + idx);     // logits for value  0
  v4f l2 = *(const v4f*)(wl + 2 * (size_t)total + idx); // logits for value +1
  v4h om, ov;
#pragma unroll
  for (int j = 0; j < 4; ++j) {
    float m  = fmaxf(l0[j], fmaxf(l1[j], l2[j]));
    float e0 = __expf(l0[j] - m);
    float e1 = __expf(l1[j] - m);
    float e2 = __expf(l2[j] - m);
    float inv  = 1.0f / (e0 + e1 + e2);
    float mean = (e2 - e0) * inv;
    float sq   = (e2 + e0) * inv;
    float var  = sq - mean * mean;
    om[j] = (_Float16)mean;
    ov[j] = (_Float16)var;
  }
  *(v4h*)(wm + idx) = om;
  *(v4h*)(wv + idx) = ov;
}

// ---------------------------------------------------------------------------
// Kernel 3: fused dual GEMM + bias epilogue.
// Block = 256 threads = 8 waves arranged 4(M) x 2(N): block tile 128x128.
// Wave tile: 32 rows x 64 cols = 2 M-tiles x 4 N-tiles x {mean,var}
//   -> 16 WMMAs per K-step vs 20 b128 loads (1.25 loads/WMMA).
//
// A layout (16-bit 16x32): lane L holds row M=L&15;
//   lanes 0-15 carry K k..k+15, lanes 16-31 carry K k+16..k+31 (matches B).
// B layout (32x16): lane L holds col N=L&15; lanes 0-15 carry K k..k+15,
//   lanes 16-31 carry K k+16..k+31 (16 contiguous f16 = 32B per lane).
// C layout: VGPR r holds M=r (lanes 0-15) / M=r+8 (lanes 16-31), N=lane&15.
// ---------------------------------------------------------------------------
__global__ __launch_bounds__(256) void k_gemm(const _Float16* __restrict__ xh,
                                              const _Float16* __restrict__ wm,
                                              const _Float16* __restrict__ wv,
                                              const float* __restrict__ blog,
                                              float* __restrict__ out) {
  const int lane  = threadIdx.x & 31;
  const int wave  = threadIdx.x >> 5;
  const int hi    = lane >> 4;            // 0: low K-half, 1: high K-half
  const int lmod  = lane & 15;
  const int waveM = wave & 3;             // 4 waves stacked in M
  const int waveN = wave >> 2;            // 2 waves stacked in N
  const int rowBase = blockIdx.y * 128 + waveM * 32;   // 32 rows per wave
  const int colBase = blockIdx.x * 128 + waveN * 64;   // 64 cols per wave

  const _Float16* aRow0 = xh + (size_t)(rowBase + lmod) * IN;       // M-tile 0
  const _Float16* aRow1 = xh + (size_t)(rowBase + 16 + lmod) * IN;  // M-tile 1

  v8f accM[2][4] = {{v8f{}, v8f{}, v8f{}, v8f{}}, {v8f{}, v8f{}, v8f{}, v8f{}}};
  v8f accV[2][4] = {{v8f{}, v8f{}, v8f{}, v8f{}}, {v8f{}, v8f{}, v8f{}, v8f{}}};

  for (int kk = 0; kk < IN; kk += 32) {
    // ---- A fragments for both M-tiles; x^2 squared in-register (f16) ----
    v8h a0lo = *(const v8h*)(aRow0 + kk + hi * 16);
    v8h a0hi = *(const v8h*)(aRow0 + kk + hi * 16 + 8);
    v8h a1lo = *(const v8h*)(aRow1 + kk + hi * 16);
    v8h a1hi = *(const v8h*)(aRow1 + kk + hi * 16 + 8);
    v16h a0 = __builtin_shufflevector(a0lo, a0hi,
                                      0, 1, 2, 3, 4, 5, 6, 7,
                                      8, 9, 10, 11, 12, 13, 14, 15);
    v16h a1 = __builtin_shufflevector(a1lo, a1hi,
                                      0, 1, 2, 3, 4, 5, 6, 7,
                                      8, 9, 10, 11, 12, 13, 14, 15);
    v16h s0 = a0 * a0;
    v16h s1 = a1 * a1;

    // near-cache prefetch of next K-tile of A
    __builtin_prefetch(aRow0 + kk + 32, 0, 3);
    __builtin_prefetch(aRow1 + kk + 32, 0, 3);

#pragma unroll
    for (int nt = 0; nt < 4; ++nt) {
      const int col = colBase + nt * 16 + lmod;
      const size_t boff = (size_t)col * IN + kk + hi * 16;
      v16h bm = *(const v16h*)(wm + boff);
      v16h bv = *(const v16h*)(wv + boff);
      __builtin_prefetch(wm + boff + 32, 0, 3);
      __builtin_prefetch(wv + boff + 32, 0, 3);
      accM[0][nt] = __builtin_amdgcn_wmma_f32_16x16x32_f16(
          false, a0, false, bm, (short)0, accM[0][nt], false, false);
      accM[1][nt] = __builtin_amdgcn_wmma_f32_16x16x32_f16(
          false, a1, false, bm, (short)0, accM[1][nt], false, false);
      accV[0][nt] = __builtin_amdgcn_wmma_f32_16x16x32_f16(
          false, s0, false, bv, (short)0, accV[0][nt], false, false);
      accV[1][nt] = __builtin_amdgcn_wmma_f32_16x16x32_f16(
          false, s1, false, bv, (short)0, accV[1][nt], false, false);
    }
  }

  // ---- epilogue: bias distribution moments + stores ----
#pragma unroll
  for (int nt = 0; nt < 4; ++nt) {
    const int col = colBase + nt * 16 + lmod;
    float l0 = blog[col];
    float l1 = blog[OUT + col];
    float l2 = blog[2 * OUT + col];
    float m  = fmaxf(l0, fmaxf(l1, l2));
    float e0 = __expf(l0 - m);
    float e1 = __expf(l1 - m);
    float e2 = __expf(l2 - m);
    float inv = 1.0f / (e0 + e1 + e2);
    float bmean = (e2 - e0) * inv;
    float bvar  = (e2 + e0) * inv - bmean * bmean;
#pragma unroll
    for (int mt = 0; mt < 2; ++mt) {
#pragma unroll
      for (int r = 0; r < 8; ++r) {
        const int row = rowBase + mt * 16 + r + hi * 8;
        const size_t o = (size_t)row * (2 * OUT) + col;
        out[o]       = accM[mt][nt][r] + bmean;   // (n, 0, out) : mean
        out[o + OUT] = accV[mt][nt][r] + bvar;    // (n, 1, out) : var
      }
    }
  }
}

// ---------------------------------------------------------------------------
// Launcher. d_in: [x (n*IN f32), W_logits (3*OUT*IN f32), b_logits (3*OUT f32)]
// d_out: (n, 2, OUT) f32. Workspace: xh (n*IN f16) | wm (OUT*IN f16) | wv.
// ---------------------------------------------------------------------------
extern "C" void kernel_launch(void* const* d_in, const int* in_sizes, int n_in,
                              void* d_out, int out_size, void* d_ws, size_t ws_size,
                              hipStream_t stream) {
  const float* x  = (const float*)d_in[0];
  const float* wl = (const float*)d_in[1];
  const float* bl = (const float*)d_in[2];
  float* out = (float*)d_out;

  const int n = in_sizes[0] / IN;             // 2048
  const int wTotal = OUT * IN;                // 16M

  char* ws = (char*)d_ws;
  size_t xhBytes = ((size_t)n * IN * sizeof(_Float16) + 255) & ~(size_t)255;
  _Float16* xh = (_Float16*)ws;
  _Float16* wmh = (_Float16*)(ws + xhBytes);
  _Float16* wvh = wmh + (size_t)wTotal;

  // 1) x -> f16
  {
    int total = n * IN;
    int threads = total / 8;
    k_cvt_x<<<(threads + 255) / 256, 256, 0, stream>>>(x, xh, total);
  }
  // 2) W_logits -> (w_mean, w_var) f16
  {
    int threads = wTotal / 4;
    k_prep_w<<<(threads + 255) / 256, 256, 0, stream>>>(wl, wmh, wvh, wTotal);
  }
  // 3) dual WMMA GEMM + bias (block tile 128x128, wave tile 32x64)
  {
    dim3 grid(OUT / 128, n / 128);
    k_gemm<<<grid, 256, 0, stream>>>(xh, wmh, wvh, bl, out);
  }
}
